// ScaleDotProductAttention_16140487098988
// MI455X (gfx1250) — matrix operand order repaired and checked by reading, not compiled, not run on hardware
//
#include <hip/hip_runtime.h>
#include <hip/hip_bf16.h>

// Problem constants (from reference): B=4, H=16, S=2048, D=64, fp32 in/out.
constexpr int BB = 4;
constexpr int HH = 16;
constexpr int SS = 2048;
constexpr int DD = 64;
constexpr float SCALE = 0.125f;    // 1/sqrt(64)
constexpr int VSTRIDE = 68;        // padded f32 row stride for staged V (16B-aligned rows, bank-spread)

typedef __attribute__((ext_vector_type(16))) _Float16 v16h;
typedef __attribute__((ext_vector_type(8)))  float    v8f;

union AFrag { v16h v; _Float16 e[16]; };

// A-matrix (16-bit, 16x32) per-lane K index for element e (ISA 7.12.2):
// lanes 0-15: e0..7 -> K0..7, e8..15 -> K16..23 ; lanes 16-31: +8
__device__ __forceinline__ int kmapA(int e, int lane) {
    return (e & 7) + ((e >> 3) << 4) + ((lane & 16) ? 8 : 0);
}
// Contiguous per-lane K order (not the ISA layout; unused here): lanes 0-15 hold K0..15,
// lanes 16-31 hold K16..31 (column N = lane & 15).
__device__ __forceinline__ int kmapB(int e, int lane) {
    return e + ((lane & 16) ? 16 : 0);
}

__global__ __launch_bounds__(128)
void sdpa_kernel(const float* __restrict__ Q,
                 const float* __restrict__ K,
                 const float* __restrict__ V,
                 const unsigned char* __restrict__ mask,  // bool [B,1,S,S], true = masked
                 float* __restrict__ ctx,                 // [B,H,S,D]
                 float* __restrict__ wout)                // [B,H,S,S]
{
    extern __shared__ float smem[];
    float*     sc   = smem;                         // 16 x S f32 scores/exp values (128 KB)
    float*     wred = smem + 16 * SS;               // 8-entry reduction buffer
    _Float16*  wf16 = (_Float16*)(smem + 16 * SS + 8);  // weights packed in WMMA-A order (64 KB)
    float*     vbuf = (float*)(wf16 + 64 * 16 * 32);    // 2 x (32 x VSTRIDE) staged V (17 KB)

    const int tid  = threadIdx.x;
    const int wid  = tid >> 5;
    const int lane = tid & 31;
    const int ln16 = lane & 15;

    const int qt = blockIdx.x;       // query tile (16 rows)
    const int bh = blockIdx.y;       // fused b*H + h
    const int b  = bh / HH;

    const float* Qb = Q + ((size_t)bh * SS + (size_t)qt * 16) * DD;
    const float* Kb = K + (size_t)bh * SS * DD;
    const float* Vb = V + (size_t)bh * SS * DD;
    const unsigned char* Mb = mask + ((size_t)b * SS + (size_t)qt * 16) * SS;

    // ---- Load Q tile as two A fragments (D=64 -> K chunks [0,32) and [32,64)) ----
    AFrag aQ0, aQ1;
    {
        const float* qrow = Qb + (size_t)ln16 * DD;
#pragma unroll
        for (int e = 0; e < 16; ++e) {
            aQ0.e[e] = (_Float16)qrow[kmapA(e, lane)];
            aQ1.e[e] = (_Float16)qrow[32 + kmapA(e, lane)];
        }
    }

    // ---- Phase 1: scores = (Q K^T) * scale into LDS, tiles split across waves ----
    for (int t = wid; t < SS / 16; t += 4) {
        const int kbase = t * 16;
        const float* krow = Kb + (size_t)(kbase + ln16) * DD;  // lane holds column N of B
        AFrag b0, b1;
#pragma unroll
        for (int e = 0; e < 16; ++e) {
            b0.e[e] = (_Float16)krow[kmapA(e, lane)];
            b1.e[e] = (_Float16)krow[32 + kmapA(e, lane)];
        }
        v8f acc = {};
        acc = __builtin_amdgcn_wmma_f32_16x16x32_f16(false, aQ0.v, false, b0.v,
                                                     (short)0, acc, false, false);
        acc = __builtin_amdgcn_wmma_f32_16x16x32_f16(false, aQ1.v, false, b1.v,
                                                     (short)0, acc, false, false);
        const int m0 = (lane & 16) ? 8 : 0;  // C/D: VGPR r -> row m0+r, N = lane&15
#pragma unroll
        for (int r = 0; r < 8; ++r)
            sc[(m0 + r) * SS + kbase + ln16] = acc[r] * SCALE;
    }
    __syncthreads();

    // ---- Phase 2: per-row mask + softmax; stream weights to global, pack f16 A-frags ----
    for (int r = 0; r < 16; ++r) {
        float* row = sc + r * SS;
        const unsigned char* mrow = Mb + (size_t)r * SS;

        // mask + row max
        float mx = -3.0e38f;
        for (int j = tid; j < SS; j += 128) {
            float s = mrow[j] ? -1.0e9f : row[j];
            row[j] = s;
            mx = fmaxf(mx, s);
        }
#pragma unroll
        for (int off = 16; off > 0; off >>= 1) mx = fmaxf(mx, __shfl_xor(mx, off, 32));
        if (lane == 0) wred[wid] = mx;
        __syncthreads();
        mx = fmaxf(fmaxf(wred[0], wred[1]), fmaxf(wred[2], wred[3]));
        __syncthreads();

        // exp + row sum
        float sm = 0.0f;
        for (int j = tid; j < SS; j += 128) {
            float e = __expf(row[j] - mx);
            row[j] = e;
            sm += e;
        }
#pragma unroll
        for (int off = 16; off > 0; off >>= 1) sm += __shfl_xor(sm, off, 32);
        if (lane == 0) wred[wid] = sm;
        __syncthreads();
        sm = wred[0] + wred[1] + wred[2] + wred[3];
        const float inv = 1.0f / sm;
        __syncthreads();

        // normalize; coalesced global store + f16 store in WMMA-A fragment order
        float* wrow = wout + ((size_t)bh * SS + (size_t)qt * 16 + r) * SS;
        for (int j = tid; j < SS; j += 128) {
            float w = row[j] * inv;
            wrow[j] = w;
            const int kt   = j >> 5;            // 32-wide K chunk
            const int kk   = j & 31;            // K within chunk
            const int half = (kk >> 3) & 1;     // which lane-half consumes it
            const int e    = (kk & 7) + ((kk & 16) ? 8 : 0);
            wf16[((kt * 16 + r) << 5) + (half << 4) + e] = (_Float16)w;
        }
    }
    __syncthreads();

    // ---- Phase 3: context = W @ V ; async-staged V chunks, each wave owns 16 columns ----
    const int col0 = wid * 16;

    auto stage = [&](int kt, int bufIdx) {
        float* dst = vbuf + bufIdx * (32 * VSTRIDE);
#pragma unroll
        for (int i = 0; i < 4; ++i) {
            const int idx = tid + 128 * i;      // 512 x 16B transfers = 32x64 f32
            const int rowi = idx >> 4;
            const int seg  = idx & 15;
            unsigned ldsOff = (unsigned)((const char*)(dst + rowi * VSTRIDE + seg * 4) -
                                         (const char*)smem);
            unsigned gOff   = (unsigned)(((kt * 32 + rowi) * DD + seg * 4) * sizeof(float));
            // CDNA5 async copy, GVS mode: LDS-addr VGPR, 32-bit offset VGPR, SGPR64 base.
            asm volatile("global_load_async_to_lds_b128 %0, %1, %2 offset:0"
                         :: "v"(ldsOff), "v"(gOff), "s"(Vb) : "memory");
        }
    };

    stage(0, 0);
    v8f acc = {};
    for (int kt = 0; kt < SS / 32; ++kt) {
        asm volatile("s_wait_asynccnt 0x0" ::: "memory");  // this wave's staged chunk landed
        __syncthreads();                                   // all waves' chunk data visible
        if (kt + 1 < SS / 32) stage(kt + 1, (kt + 1) & 1); // overlap next copy with WMMA

        // A: one aligned 32B vector load of the pre-packed f16 weights fragment
        AFrag a;
        a.v = *(const v16h*)(wf16 + (((kt * 16 + ln16) << 5) + (lane & 16)));
        // B: V columns from the staged LDS chunk (2-way max bank conflict)
        const float* vb = vbuf + (kt & 1) * (32 * VSTRIDE);
        AFrag bv;
#pragma unroll
        for (int e = 0; e < 16; ++e)
            bv.e[e] = (_Float16)vb[kmapA(e, lane) * VSTRIDE + col0 + ln16];

        acc = __builtin_amdgcn_wmma_f32_16x16x32_f16(false, a.v, false, bv.v,
                                                     (short)0, acc, false, false);
        __syncthreads();   // reads of this buffer done before it is restaged
    }

    const int m0 = (lane & 16) ? 8 : 0;
#pragma unroll
    for (int r = 0; r < 8; ++r)
        ctx[((size_t)bh * SS + (size_t)qt * 16 + m0 + r) * DD + col0 + ln16] = acc[r];
}

extern "C" void kernel_launch(void* const* d_in, const int* in_sizes, int n_in,
                              void* d_out, int out_size, void* d_ws, size_t ws_size,
                              hipStream_t stream) {
    (void)in_sizes; (void)n_in; (void)d_ws; (void)ws_size; (void)out_size;

    const float* Q = (const float*)d_in[0];
    const float* K = (const float*)d_in[1];
    const float* V = (const float*)d_in[2];
    const unsigned char* mask = (const unsigned char*)d_in[3];  // jnp.bool_ -> 1 byte/elem

    float* ctx  = (float*)d_out;                                  // first output
    float* wout = ctx + (size_t)BB * HH * SS * DD;                // second output

    dim3 grid(SS / 16, BB * HH);
    dim3 block(128);
    size_t smemBytes = (size_t)(16 * SS + 8) * sizeof(float)      // scores + reduction
                     + (size_t)(64 * 16 * 32) * sizeof(_Float16)  // packed f16 weights
                     + (size_t)(2 * 32 * VSTRIDE) * sizeof(float);// double-buffered V
    sdpa_kernel<<<grid, block, smemBytes, stream>>>(Q, K, V, mask, ctx, wout);
}
